// GNN_73658689126419
// MI455X (gfx1250) — compile-verified, matrix operand order checked by reading, not run on hardware
//
#include <hip/hip_runtime.h>

// ---------------------------------------------------------------------------
// CDNA5 (gfx1250) GCN pipeline: bf16 WMMA GEMMs (row-stripe A-reuse) +
// L2-resident edge scatter with f32 agent-scope atomics.
// ---------------------------------------------------------------------------

typedef __attribute__((ext_vector_type(16))) __bf16 v16bf;
typedef __attribute__((ext_vector_type(8)))  float  v8f;

#define F128 128

// ---- Weight packing: f32 row-major W[k][n] -> bf16 WMMA B-fragment layout ----
// Fragment storage: Wpk[((kt*ntiles + nt)*32 + lane)*16 + i]
// Lane L holds column N = nt*16 + (L&15); element i holds
// K = kt*32 + (i<8 ? i : i+8) + (L>=16 ? 8 : 0)   (per ISA 16-bit B layout)
__global__ void pack_w_bf16(const float* __restrict__ W, __bf16* __restrict__ Wpk,
                            int ntiles, int ncols, int total) {
    int tid = blockIdx.x * blockDim.x + threadIdx.x;
    if (tid >= total) return;
    int i    = tid & 15;
    int lane = (tid >> 4) & 31;
    int nt   = (tid >> 9) % ntiles;
    int kt   = tid / (512 * ntiles);
    int K    = kt * 32 + ((i < 8) ? i : i + 8) + ((lane >= 16) ? 8 : 0);
    int col  = nt * 16 + (lane & 15);
    Wpk[tid] = (__bf16)W[(size_t)K * ncols + col];
}

// ---- GEMM: C[M x ncols] = A[M x 128] @ W (+bias).
// One wave per 16-row stripe: A fragment built once per K-tile and reused
// across all NT column tiles (NT accumulators live in registers).
template <int NT>
__global__ void wmma_gemm_stripe(const float* __restrict__ A, const __bf16* __restrict__ Wpk,
                                 const float* __restrict__ bias, float* __restrict__ C,
                                 int mtiles, int ncols) {
    int wid  = (blockIdx.x * blockDim.x + threadIdx.x) >> 5;
    int lane = threadIdx.x & 31;
    if (wid >= mtiles) return;
    int lrow = lane & 15;
    int hi   = lane >> 4;                       // 0: K-half 0, 1: K-half 1

    const float* arow = A + (size_t)(wid * 16 + lrow) * F128 + (hi ? 8 : 0);
    const v16bf* bp   = (const v16bf*)Wpk;

    v8f acc[NT];
#pragma unroll
    for (int nt = 0; nt < NT; ++nt) acc[nt] = (v8f){0.f,0.f,0.f,0.f,0.f,0.f,0.f,0.f};

#pragma unroll
    for (int kt = 0; kt < 4; ++kt) {
        const float* ap = arow + kt * 32;
        float4 f0 = *(const float4*)(ap);
        float4 f1 = *(const float4*)(ap + 4);
        float4 f2 = *(const float4*)(ap + 16);
        float4 f3 = *(const float4*)(ap + 20);
        v16bf a;
        a[0]=(__bf16)f0.x; a[1]=(__bf16)f0.y; a[2]=(__bf16)f0.z; a[3]=(__bf16)f0.w;
        a[4]=(__bf16)f1.x; a[5]=(__bf16)f1.y; a[6]=(__bf16)f1.z; a[7]=(__bf16)f1.w;
        a[8]=(__bf16)f2.x; a[9]=(__bf16)f2.y; a[10]=(__bf16)f2.z; a[11]=(__bf16)f2.w;
        a[12]=(__bf16)f3.x; a[13]=(__bf16)f3.y; a[14]=(__bf16)f3.z; a[15]=(__bf16)f3.w;
#pragma unroll
        for (int nt = 0; nt < NT; ++nt) {
            v16bf b = bp[(size_t)(kt * NT + nt) * 32 + lane];
            acc[nt] = __builtin_amdgcn_wmma_f32_16x16x32_bf16(
                /*neg_a=*/false, a, /*neg_b=*/false, b,
                /*c_mod=*/(short)0, acc[nt], /*reuse_a=*/false, /*reuse_b=*/false);
        }
    }

    float* crow = C + (size_t)(wid * 16 + hi * 8) * ncols + lrow;
#pragma unroll
    for (int nt = 0; nt < NT; ++nt) {
        int col  = nt * 16;
        float bv = bias ? bias[col + lrow] : 0.f;
#pragma unroll
        for (int j = 0; j < 8; ++j) crow[(size_t)j * ncols + col] = acc[nt][j] + bv;
    }
}

// ---- Degree / normalization ----
__global__ void set_const_f32(float* p, float v, int n) {
    int t = blockIdx.x * blockDim.x + threadIdx.x;
    if (t < n) p[t] = v;
}
__global__ void deg_count(const int* __restrict__ ei, float* __restrict__ deg, int E) {
    int e = blockIdx.x * blockDim.x + threadIdx.x;
    if (e >= E) return;
    int d = ei[E + e];   // dst row
    __hip_atomic_fetch_add(&deg[d], 1.0f, __ATOMIC_RELAXED, __HIP_MEMORY_SCOPE_AGENT);
}
__global__ void deg_finalize(float* __restrict__ degio /*in: deg, out: nself*/,
                             float* __restrict__ dinv, int n) {
    int t = blockIdx.x * blockDim.x + threadIdx.x;
    if (t >= n) return;
    float d = degio[t];
    dinv[t]  = rsqrtf(d);
    degio[t] = 1.0f / d;      // norm_self
}

// ---- Zero a float buffer (float4 stores) ----
__global__ void zero_f4(float4* p, int n4) {
    int t = blockIdx.x * blockDim.x + threadIdx.x;
    if (t < n4) p[t] = make_float4(0.f, 0.f, 0.f, 0.f);
}

// ---- Edge scatter: AGG[dst] += H[src] * dinv[src]*dinv[dst].  Wave per edge. ----
__global__ void scatter_edges(const int* __restrict__ ei, const float* __restrict__ H,
                              const float* __restrict__ dinv, float* __restrict__ AGG,
                              int E) {
    int tid = blockIdx.x * blockDim.x + threadIdx.x;
    int e   = tid >> 5;
    if (e >= E) return;
    int c0 = (tid & 31) << 2;               // 4 consecutive features per lane
    int s  = ei[e];
    int d  = ei[E + e];
    if (e + 1 < E)                          // gfx1250 global_prefetch of next gather row
        __builtin_prefetch(H + (size_t)ei[e + 1] * F128 + c0, 0, 1);
    float w = dinv[s] * dinv[d];
    float4 h4 = *(const float4*)(H + (size_t)s * F128 + c0);
    float* ap = AGG + (size_t)d * F128 + c0;
    __hip_atomic_fetch_add(ap + 0, h4.x * w, __ATOMIC_RELAXED, __HIP_MEMORY_SCOPE_AGENT);
    __hip_atomic_fetch_add(ap + 1, h4.y * w, __ATOMIC_RELAXED, __HIP_MEMORY_SCOPE_AGENT);
    __hip_atomic_fetch_add(ap + 2, h4.z * w, __ATOMIC_RELAXED, __HIP_MEMORY_SCOPE_AGENT);
    __hip_atomic_fetch_add(ap + 3, h4.w * w, __ATOMIC_RELAXED, __HIP_MEMORY_SCOPE_AGENT);
}

// ---- Combine: out = relu(AGG + H*nself + bias [+ xres]) ----
__global__ void combine_relu(const float* __restrict__ AGG, const float* __restrict__ H,
                             const float* __restrict__ nself, const float* __restrict__ bias,
                             const float* __restrict__ xres, float* __restrict__ out,
                             int total) {
    int t = blockIdx.x * blockDim.x + threadIdx.x;
    if (t >= total) return;
    int row = t >> 7, col = t & 127;
    float v = AGG[t] + H[t] * nself[row] + bias[col];
    if (xres) v += xres[t];
    out[t] = fmaxf(v, 0.f);
}

// ---------------------------------------------------------------------------
extern "C" void kernel_launch(void* const* d_in, const int* in_sizes, int n_in,
                              void* d_out, int out_size, void* d_ws, size_t ws_size,
                              hipStream_t stream) {
    const float* x    = (const float*)d_in[0];
    const int*   ei   = (const int*)d_in[1];     // [2, E] (src row then dst row)
    const float* W1   = (const float*)d_in[2];
    const float* b1   = (const float*)d_in[3];
    const float* W2   = (const float*)d_in[4];
    const float* b2   = (const float*)d_in[5];
    const float* W3   = (const float*)d_in[6];
    const float* b3   = (const float*)d_in[7];
    const float* Wres = (const float*)d_in[8];
    const float* bres = (const float*)d_in[9];
    const float* Wlin = (const float*)d_in[10];
    const float* blin = (const float*)d_in[11];
    float* out = (float*)d_out;

    const int N   = in_sizes[0] / F128;          // 50000 (multiple of 16)
    const int E   = in_sizes[1] / 2;             // 640000
    const int OUT = 64;
    const int MT  = N / 16;                      // row tiles
    const int NF  = N * F128;

    // ---- workspace carve-up (256B aligned) ----
    char* base = (char*)d_ws;
    size_t off = 0;
    auto carve = [&](size_t bytes) -> char* {
        char* p = base + off;
        off = (off + bytes + 255) & ~(size_t)255;
        return p;
    };
    float*  dinv  = (float*) carve((size_t)N * 4);
    float*  nself = (float*) carve((size_t)N * 4);   // holds deg first, then 1/deg
    __bf16* pW1   = (__bf16*)carve(128 * 128 * 2);
    __bf16* pW2   = (__bf16*)carve(128 * 128 * 2);
    __bf16* pW3   = (__bf16*)carve(128 * 128 * 2);
    __bf16* pWres = (__bf16*)carve(128 * 128 * 2);
    __bf16* pWlin = (__bf16*)carve(128 * 64  * 2);
    float*  ACT   = (float*) carve((size_t)NF * 4);  // x_res, then ping activation
    float*  Hf    = (float*) carve((size_t)NF * 4);  // GEMM output h
    float*  AGG   = (float*) carve((size_t)NF * 4);  // scatter accumulator
    (void)ws_size; (void)n_in; (void)out_size;

    const int TPB = 256;

    // 1) pack weights to WMMA bf16 fragments
    pack_w_bf16<<<(16384 + TPB - 1) / TPB, TPB, 0, stream>>>(W1,   pW1,   8, 128, 16384);
    pack_w_bf16<<<(16384 + TPB - 1) / TPB, TPB, 0, stream>>>(W2,   pW2,   8, 128, 16384);
    pack_w_bf16<<<(16384 + TPB - 1) / TPB, TPB, 0, stream>>>(W3,   pW3,   8, 128, 16384);
    pack_w_bf16<<<(16384 + TPB - 1) / TPB, TPB, 0, stream>>>(Wres, pWres, 8, 128, 16384);
    pack_w_bf16<<<(8192  + TPB - 1) / TPB, TPB, 0, stream>>>(Wlin, pWlin, 4, 64,  8192);

    // 2) degrees -> dinv, nself
    set_const_f32<<<(N + TPB - 1) / TPB, TPB, 0, stream>>>(nself, 1.0f, N);  // self-loop
    deg_count<<<(E + TPB - 1) / TPB, TPB, 0, stream>>>(ei, nself, E);
    deg_finalize<<<(N + TPB - 1) / TPB, TPB, 0, stream>>>(nself, dinv, N);

    // 3) x_res = x @ Wres + bres  -> ACT
    int gw = (MT + 7) / 8;          // 8 row-stripe waves per 256-thread block
    wmma_gemm_stripe<8><<<gw, TPB, 0, stream>>>(x, pWres, bres, ACT, MT, 128);

    // 4) three GCN layers
    const __bf16* pw[3]  = {pW1, pW2, pW3};
    const float*  bb[3]  = {b1, b2, b3};
    for (int layer = 0; layer < 3; ++layer) {
        const float* act_in = (layer == 0) ? x : ACT;
        wmma_gemm_stripe<8><<<gw, TPB, 0, stream>>>(act_in, pw[layer], nullptr, Hf, MT, 128);
        zero_f4<<<(NF / 4 + TPB - 1) / TPB, TPB, 0, stream>>>((float4*)AGG, NF / 4);
        scatter_edges<<<(E * 32 + TPB - 1) / TPB, TPB, 0, stream>>>(ei, Hf, dinv, AGG, E);
        // layer 0: add residual (reads & overwrites ACT elementwise, safe)
        combine_relu<<<(NF + TPB - 1) / TPB, TPB, 0, stream>>>(
            AGG, Hf, nself, bb[layer], (layer == 0) ? ACT : nullptr, ACT, NF);
    }

    // 5) out = ACT @ Wlin + blin
    wmma_gemm_stripe<4><<<gw, TPB, 0, stream>>>(ACT, pWlin, blin, out, MT, OUT);
}
